// CombinedModel_sink_feature_multimodel_70437463655094
// MI455X (gfx1250) — matrix operand order repaired and checked by reading, not compiled, not run on hardware
//
#include <hip/hip_runtime.h>
#include <math.h>

// ---------------------------------------------------------------------------
// MI455X (gfx1250) fused implementation, round 3.
//  - GEMMs: v_wmma_f32_16x16x32_bf16, 32x32 wave tiles (4 acc, 2x2 frag reuse),
//    software-pipelined k-loop (next fragments loaded before current WMMAs).
//  - Affinity rewritten as F[b] @ (M_sym @ text1^T): 20x FLOP reduction.
//  - Per-batch branch kernel keeps s (333x28) in LDS through instance-norm,
//    100 Sinkhorn iterations, exp, feats GEMM and concat; A-fragment loads
//    now pipelined against the WMMA pairs as well.
// ---------------------------------------------------------------------------

typedef __attribute__((ext_vector_type(16))) __bf16        v16bf;
typedef __attribute__((ext_vector_type(8)))  float         v8f;
typedef __attribute__((ext_vector_type(4)))  unsigned int  u32x4;

#define BB      128
#define NN      333
#define DIN     333
#define KPAD1   352          // DIN padded to multiple of 32
#define DH      1024
#define DD      512
#define TT      28
#define TPAD    32
#define MROWS   (BB * NN)    // 42624 = 333 * 128
#define SINK_IT 100

union FragBF {
  v16bf v;
  u32x4 q[2];
};

__device__ __forceinline__ unsigned short f32_to_bf16u(float x) {
  unsigned int u = __float_as_uint(x);
  unsigned int r = u + 0x7FFFu + ((u >> 16) & 1u);
  return (unsigned short)(r >> 16);
}
__device__ __forceinline__ float bf16u_to_f32(unsigned short h) {
  return __uint_as_float(((unsigned int)h) << 16);
}
__device__ __forceinline__ float gelu_tanh(float x) {
  float x3 = x * x * x;
  return 0.5f * x * (1.0f + tanhf(0.7978845608028654f * (x + 0.044715f * x3)));
}

// A fragment (16x32 bf16, MxK): lane L -> row base + (L&15);
// elems 0..7 = K[k0 + (L>>4)*8 ..], elems 8..15 = K[k0+16 + (L>>4)*8 ..].
__device__ __forceinline__ v16bf load_a_frag(const unsigned short* __restrict__ A,
                                             int row, int ldk, int k0, int lane) {
  FragBF f;
  const unsigned short* p = A + (size_t)row * ldk + k0 + ((lane >> 4) << 3);
  f.q[0] = *reinterpret_cast<const u32x4*>(p);
  f.q[1] = *reinterpret_cast<const u32x4*>(p + 16);
  return f.v;
}

// B fragment (32x16 bf16, KxN) from an NxK row-major weight:
// lane L -> col base + (L&15); elems 0..7 = K[k0 + (L>>4)*8 ..], elems 8..15 = K[k0+16 + (L>>4)*8 ..].
__device__ __forceinline__ v16bf load_b_frag(const unsigned short* __restrict__ Bw,
                                             int col, int ldk, int k0, int lane) {
  FragBF f;
  const unsigned short* p = Bw + (size_t)col * ldk + k0 + ((lane >> 4) << 3);
  f.q[0] = *reinterpret_cast<const u32x4*>(p);
  f.q[1] = *reinterpret_cast<const u32x4*>(p + 16);
  return f.v;
}

__device__ __forceinline__ v8f wmma_bf16(v16bf a, v16bf b, v8f c) {
  return __builtin_amdgcn_wmma_f32_16x16x32_bf16(false, a, false, b,
                                                 (short)0, c, false, false);
}

// ---------------------------------------------------------------------------
// WMMA GEMM: C(MxN,bf16) = epi(A(MxKpad) * B(NxKpad)^T + bias), Kpad=KSTEPS*32.
// 256 threads = 8 waves (4x2); each wave owns a 32x32 tile -> block 128x64.
// Software-pipelined: fragments for k-step s+1 are in flight during the four
// WMMAs of step s.
// ---------------------------------------------------------------------------
template <bool EPI_GELU, int KSTEPS>
__global__ void __launch_bounds__(256)
wmma_gemm_bf16(const unsigned short* __restrict__ A,
               const unsigned short* __restrict__ Bw,
               const float* __restrict__ bias,
               unsigned short* __restrict__ C,
               int N) {
  constexpr int Kpad = KSTEPS * 32;
  const int tid  = threadIdx.x;
  const int lane = tid & 31;
  const int wave = tid >> 5;
  const int m0 = (blockIdx.x * 4 + (wave & 3)) * 32;
  const int n0 = (blockIdx.y * 2 + (wave >> 2)) * 32;

  const int ar0 = m0 + (lane & 15);
  const int ar1 = ar0 + 16;
  const int bc0 = n0 + (lane & 15);
  const int bc1 = bc0 + 16;

  v8f c00 = {}, c01 = {}, c10 = {}, c11 = {};

  v16bf a0 = load_a_frag(A, ar0, Kpad, 0, lane);
  v16bf a1 = load_a_frag(A, ar1, Kpad, 0, lane);
  v16bf b0 = load_b_frag(Bw, bc0, Kpad, 0, lane);
  v16bf b1 = load_b_frag(Bw, bc1, Kpad, 0, lane);

#pragma unroll
  for (int s = 0; s < KSTEPS - 1; ++s) {
    const int kn = (s + 1) * 32;
    v16bf na0 = load_a_frag(A, ar0, Kpad, kn, lane);
    v16bf na1 = load_a_frag(A, ar1, Kpad, kn, lane);
    v16bf nb0 = load_b_frag(Bw, bc0, Kpad, kn, lane);
    v16bf nb1 = load_b_frag(Bw, bc1, Kpad, kn, lane);
    c00 = wmma_bf16(a0, b0, c00);
    c01 = wmma_bf16(a0, b1, c01);
    c10 = wmma_bf16(a1, b0, c10);
    c11 = wmma_bf16(a1, b1, c11);
    a0 = na0; a1 = na1; b0 = nb0; b1 = nb1;
  }
  c00 = wmma_bf16(a0, b0, c00);
  c01 = wmma_bf16(a0, b1, c01);
  c10 = wmma_bf16(a1, b0, c10);
  c11 = wmma_bf16(a1, b1, c11);

  const float bv0 = bias[bc0];
  const float bv1 = bias[bc1];
  const int mb0 = m0 + ((lane >> 4) << 3);        // rows for c00/c01
  const int mb1 = mb0 + 16;                       // rows for c10/c11
#pragma unroll
  for (int v = 0; v < 8; ++v) {
    float x00 = c00[v] + bv0, x01 = c01[v] + bv1;
    float x10 = c10[v] + bv0, x11 = c11[v] + bv1;
    if (EPI_GELU) {
      x00 = gelu_tanh(x00); x01 = gelu_tanh(x01);
      x10 = gelu_tanh(x10); x11 = gelu_tanh(x11);
    }
    C[(size_t)(mb0 + v) * N + bc0] = f32_to_bf16u(x00);
    C[(size_t)(mb0 + v) * N + bc1] = f32_to_bf16u(x01);
    C[(size_t)(mb1 + v) * N + bc0] = f32_to_bf16u(x10);
    C[(size_t)(mb1 + v) * N + bc1] = f32_to_bf16u(x11);
  }
}

// ---------------------------------------------------------------------------
// Fused branch kernel: one block per batch sample b.
//   s = F[b](333x512) @ P^T           (WMMA, result -> LDS)
//   s = InstanceNorm(s); 100x Sinkhorn; s = exp(s)
//   combine[b, m, 0:512]    = sum_n F[b,n,:] * s[n,m]
//   combine[b, m, 512:1024] = text1[m, :]
// ---------------------------------------------------------------------------
__global__ void __launch_bounds__(256)
branch_kernel(const unsigned short* __restrict__ F,   // (B*333) x 512 bf16
              const unsigned short* __restrict__ P,   // 32 x 512 bf16 (N x K)
              const float* __restrict__ text1,        // 28 x 512 f32
              const float* __restrict__ gamma_p,
              const float* __restrict__ beta_p,
              float* __restrict__ combine) {          // B x 28 x 1024
  __shared__ float s_tile[NN * TT];   // 333*28 = 9324 floats (37 KB)
  __shared__ float red[256];
  __shared__ float red2[256];

  const int b    = blockIdx.x;
  const int tid  = threadIdx.x;
  const int lane = tid & 31;
  const int wave = tid >> 5;
  const unsigned short* Fb = F + (size_t)b * NN * DD;

  // ---- WMMA affinity: 21 M-tiles over 8 waves; both N-tiles share A ----
  // A-fragment loads pipelined one k-step ahead of the WMMA pair.
  for (int tm = wave; tm < 21; tm += 8) {
    const int arow   = tm * 16 + (lane & 15);
    const int arow_c = (arow < NN) ? arow : (NN - 1);   // clamp: garbage stays in
    const int bcol0  = lane & 15;                       // dropped rows only
    const int bcol1  = bcol0 + 16;
    v8f acc0 = {}, acc1 = {};
    v16bf a = load_a_frag(Fb, arow_c, DD, 0, lane);
#pragma unroll
    for (int ks = 0; ks < (DD / 32) - 1; ++ks) {
      v16bf an = load_a_frag(Fb, arow_c, DD, (ks + 1) * 32, lane);
      v16bf p0 = load_b_frag(P, bcol0, DD, ks * 32, lane);
      v16bf p1 = load_b_frag(P, bcol1, DD, ks * 32, lane);
      acc0 = wmma_bf16(a, p0, acc0);
      acc1 = wmma_bf16(a, p1, acc1);
      a = an;
    }
    {
      v16bf p0 = load_b_frag(P, bcol0, DD, DD - 32, lane);
      v16bf p1 = load_b_frag(P, bcol1, DD, DD - 32, lane);
      acc0 = wmma_bf16(a, p0, acc0);
      acc1 = wmma_bf16(a, p1, acc1);
    }
    const int mbase = tm * 16 + ((lane >> 4) << 3);
#pragma unroll
    for (int v = 0; v < 8; ++v) {
      const int m = mbase + v;
      if (m < NN) {
        if (bcol0 < TT) s_tile[m * TT + bcol0] = acc0[v];
        if (bcol1 < TT) s_tile[m * TT + bcol1] = acc1[v];
      }
    }
  }
  __syncthreads();

  // ---- InstanceNorm over all 333*28 elements ----
  float psum = 0.0f, psq = 0.0f;
  for (int i = tid; i < NN * TT; i += 256) {
    float x = s_tile[i];
    psum += x;
    psq  += x * x;
  }
  red[tid] = psum;
  red2[tid] = psq;
  __syncthreads();
  for (int s = 128; s > 0; s >>= 1) {
    if (tid < s) { red[tid] += red[tid + s]; red2[tid] += red2[tid + s]; }
    __syncthreads();
  }
  const float inv_n = 1.0f / (float)(NN * TT);
  const float mean = red[0] * inv_n;
  const float var  = red2[0] * inv_n - mean * mean;
  const float g    = gamma_p[0];
  const float be   = beta_p[0];
  const float rstd = rsqrtf(var + 1e-5f);
  for (int i = tid; i < NN * TT; i += 256)
    s_tile[i] = g * (s_tile[i] - mean) * rstd + be;
  __syncthreads();

  // ---- log-domain Sinkhorn: row (28) then column (333) normalization ----
  for (int it = 0; it < SINK_IT; ++it) {
    for (int r = tid; r < NN; r += 256) {
      float* row = s_tile + r * TT;
      float mx = row[0];
#pragma unroll
      for (int c = 1; c < TT; ++c) mx = fmaxf(mx, row[c]);
      float sm = 0.0f;
#pragma unroll
      for (int c = 0; c < TT; ++c) sm += expf(row[c] - mx);
      const float lse = mx + logf(sm);
#pragma unroll
      for (int c = 0; c < TT; ++c) row[c] -= lse;
    }
    __syncthreads();
    if (tid < TT) {
      const int c = tid;
      float mx = s_tile[c];
      for (int r = 1; r < NN; ++r) mx = fmaxf(mx, s_tile[r * TT + c]);
      float sm = 0.0f;
      for (int r = 0; r < NN; ++r) sm += expf(s_tile[r * TT + c] - mx);
      const float lse = mx + logf(sm);
      for (int r = 0; r < NN; ++r) s_tile[r * TT + c] -= lse;
    }
    __syncthreads();
  }
  for (int i = tid; i < NN * TT; i += 256) s_tile[i] = expf(s_tile[i]);
  __syncthreads();

  // ---- feats = s^T @ F[b] (vectorized 8-wide) and concat with text1 ----
  float* out = combine + (size_t)b * TT * (2 * DD);
  for (int chunk = tid; chunk < TT * (DD / 8); chunk += 256) {
    const int m  = chunk / (DD / 8);
    const int d8 = chunk % (DD / 8);
    float acc[8] = {0, 0, 0, 0, 0, 0, 0, 0};
    const unsigned short* fp = Fb + d8 * 8;
    for (int n = 0; n < NN; ++n) {
      const u32x4 q = *reinterpret_cast<const u32x4*>(fp + (size_t)n * DD);
      const float sv = s_tile[n * TT + m];
#pragma unroll
      for (int j = 0; j < 4; ++j) {
        acc[2 * j]     += bf16u_to_f32((unsigned short)(q[j] & 0xFFFFu)) * sv;
        acc[2 * j + 1] += bf16u_to_f32((unsigned short)(q[j] >> 16)) * sv;
      }
    }
    float* op = out + (size_t)m * (2 * DD) + d8 * 8;
    const float* tp = text1 + (size_t)m * DD + d8 * 8;
#pragma unroll
    for (int j = 0; j < 8; ++j) {
      op[j]      = acc[j];
      op[DD + j] = tp[j];
    }
  }
}

// ---------------------------------------------------------------------------
// Small VALU helpers
// ---------------------------------------------------------------------------

__global__ void __launch_bounds__(256)
convert_pad_kernel(const float* __restrict__ src, unsigned short* __restrict__ dst,
                   int K, int Kpad, long long total) {
  long long idx = (long long)blockIdx.x * 256 + threadIdx.x;
  if (idx >= total) return;
  const int d = (int)(idx % Kpad);
  const long long r = idx / Kpad;
  const float v = (d < K) ? src[r * K + d] : 0.0f;
  dst[idx] = f32_to_bf16u(v);
}

__global__ void __launch_bounds__(256)
t1w_proj_kernel(const float* __restrict__ t1w, const float* __restrict__ w,
                const float* __restrict__ bvec, unsigned short* __restrict__ dst) {
  long long idx = (long long)blockIdx.x * 256 + threadIdx.x;
  const long long total = (long long)MROWS * KPAD1;
  if (idx >= total) return;
  const int d = (int)(idx % KPAD1);
  const long long r = idx / KPAD1;    // b*333 + n
  float v = 0.0f;
  if (d < DIN) {
    const float* tp = t1w + r * 9;
    const float* wp = w + (size_t)d * 9;
    float acc = bvec[d];
#pragma unroll
    for (int k = 0; k < 9; ++k) acc += tp[k] * wp[k];
    v = fmaxf(acc, 0.0f);
  }
  dst[idx] = f32_to_bf16u(v);
}

__global__ void __launch_bounds__(256)
msym_p_kernel(const float* __restrict__ A, const float* __restrict__ text1,
              unsigned short* __restrict__ P) {
  const int idx = blockIdx.x * 256 + threadIdx.x;
  if (idx >= TPAD * DD) return;
  const int t = idx >> 9;
  const int d = idx & (DD - 1);
  float acc = 0.0f;
  if (t < TT) {
    for (int e = 0; e < DD; ++e)
      acc += 0.5f * (A[(size_t)d * DD + e] + A[(size_t)e * DD + d]) *
             text1[(size_t)t * DD + e];
  }
  P[idx] = f32_to_bf16u(acc);
}

__global__ void __launch_bounds__(256)
pred_kernel(const float* __restrict__ combine, const float* __restrict__ task_w,
            const float* __restrict__ task_b, float* __restrict__ pred) {
  __shared__ float red[256];
  const int b = blockIdx.x;
  const int t = blockIdx.y;
  const int tid = threadIdx.x;
  const float* f = combine + ((size_t)b * TT + t) * (2 * DD);
  const float* w = task_w + (size_t)t * (2 * DD);
  float acc = 0.0f;
  for (int d = tid; d < 2 * DD; d += 256) acc += f[d] * w[d];
  red[tid] = acc;
  __syncthreads();
  for (int s = 128; s > 0; s >>= 1) {
    if (tid < s) red[tid] += red[tid + s];
    __syncthreads();
  }
  if (tid == 0) pred[(size_t)b * TT + t] = red[0] + task_b[t];
}

__global__ void __launch_bounds__(256)
copy_rp_kernel(const float* __restrict__ task_w, float* __restrict__ rp1,
               float* __restrict__ rp2) {
  const int i = blockIdx.x * 256 + threadIdx.x;
  if (i < TT * 2 * DD) {
    const float v = task_w[i];
    rp1[i] = v;
    rp2[i] = v;
  }
}

// ---------------------------------------------------------------------------
// Host-side launch
// ---------------------------------------------------------------------------
extern "C" void kernel_launch(void* const* d_in, const int* in_sizes, int n_in,
                              void* d_out, int out_size, void* d_ws, size_t ws_size,
                              hipStream_t stream) {
  (void)in_sizes; (void)n_in; (void)out_size; (void)ws_size;

  const float* fmri    = (const float*)d_in[0];
  const float* t1w     = (const float*)d_in[1];
  const float* text    = (const float*)d_in[2];
  const float* fc1_w   = (const float*)d_in[3];
  const float* fc1_b   = (const float*)d_in[4];
  const float* ext_w1  = (const float*)d_in[5];
  const float* ext_b1  = (const float*)d_in[6];
  const float* ext_w2  = (const float*)d_in[7];
  const float* ext_b2  = (const float*)d_in[8];
  const float* aff_A   = (const float*)d_in[9];
  const float* aff_tA  = (const float*)d_in[10];
  const float* g1      = (const float*)d_in[11];
  const float* b1s     = (const float*)d_in[12];
  const float* g2      = (const float*)d_in[13];
  const float* b2s     = (const float*)d_in[14];
  const float* task_w  = (const float*)d_in[15];
  const float* task_b  = (const float*)d_in[16];

  const float* text1 = text + (size_t)TT * DD;    // text[1]

  // Workspace carve-up (bf16 buffers, all 16B aligned)
  unsigned short* Abuf = (unsigned short*)d_ws;            // MROWS x 352
  unsigned short* Hbuf = Abuf + (size_t)MROWS * KPAD1;     // MROWS x 1024
  unsigned short* Fbuf = Hbuf + (size_t)MROWS * DH;        // MROWS x 512
  unsigned short* W1b  = Fbuf + (size_t)MROWS * DD;        // 1024 x 352
  unsigned short* W2b  = W1b + (size_t)DH * KPAD1;         // 512 x 1024
  unsigned short* Pbuf = W2b + (size_t)DD * DH;            // 32 x 512

  float* out   = (float*)d_out;
  float* pred1 = out;
  float* rp1   = pred1 + (size_t)BB * TT;
  float* comb1 = rp1 + (size_t)TT * 2 * DD;
  float* pred2 = comb1 + (size_t)BB * TT * 2 * DD;
  float* rp2   = pred2 + (size_t)BB * TT;
  float* comb2 = rp2 + (size_t)TT * 2 * DD;

  const long long totA = (long long)MROWS * KPAD1;
  const int blkA  = (int)((totA + 255) / 256);
  const int blkW1 = (int)(((long long)DH * KPAD1 + 255) / 256);
  const int blkW2 = (int)(((long long)DD * DH + 255) / 256);

  convert_pad_kernel<<<blkW1, 256, 0, stream>>>(ext_w1, W1b, DIN, KPAD1,
                                                (long long)DH * KPAD1);
  convert_pad_kernel<<<blkW2, 256, 0, stream>>>(ext_w2, W2b, DH, DH,
                                                (long long)DD * DH);

  // ===== Branch 1: fmri =====
  convert_pad_kernel<<<blkA, 256, 0, stream>>>(fmri, Abuf, DIN, KPAD1, totA);
  wmma_gemm_bf16<true, KPAD1 / 32><<<dim3(MROWS / 128, DH / 64), 256, 0, stream>>>(
      Abuf, W1b, ext_b1, Hbuf, DH);
  wmma_gemm_bf16<false, DH / 32><<<dim3(MROWS / 128, DD / 64), 256, 0, stream>>>(
      Hbuf, W2b, ext_b2, Fbuf, DD);
  msym_p_kernel<<<(TPAD * DD) / 256, 256, 0, stream>>>(aff_A, text1, Pbuf);
  branch_kernel<<<BB, 256, 0, stream>>>(Fbuf, Pbuf, text1, g1, b1s, comb1);

  // ===== Branch 2: t1w =====
  t1w_proj_kernel<<<blkA, 256, 0, stream>>>(t1w, fc1_w, fc1_b, Abuf);
  wmma_gemm_bf16<true, KPAD1 / 32><<<dim3(MROWS / 128, DH / 64), 256, 0, stream>>>(
      Abuf, W1b, ext_b1, Hbuf, DH);
  wmma_gemm_bf16<false, DH / 32><<<dim3(MROWS / 128, DD / 64), 256, 0, stream>>>(
      Hbuf, W2b, ext_b2, Fbuf, DD);
  msym_p_kernel<<<(TPAD * DD) / 256, 256, 0, stream>>>(aff_tA, text1, Pbuf);
  branch_kernel<<<BB, 256, 0, stream>>>(Fbuf, Pbuf, text1, g2, b2s, comb2);

  // ===== Heads + passthrough outputs =====
  pred_kernel<<<dim3(BB, TT), 256, 0, stream>>>(comb1, task_w, task_b, pred1);
  pred_kernel<<<dim3(BB, TT), 256, 0, stream>>>(comb2, task_w, task_b, pred2);
  copy_rp_kernel<<<(TT * 2 * DD + 255) / 256, 256, 0, stream>>>(task_w, rp1, rp2);
}
